// CSI_1812476199070
// MI455X (gfx1250) — matrix operand order repaired and checked by reading, not compiled, not run on hardware
//
#include <hip/hip_runtime.h>
#include <hip/hip_bf16.h>
#include <math.h>

#define Bb  4
#define Cc  256
#define Lt  4096      // H*W
#define DMc 64
#define DIc 128
#define NSc 16
#define KCc 4
#define NBc 16        // 4*B
#define NTt (NBc*Lt)  // 65536 rows for the mamba stream
#define EPSF 1e-5f

typedef _Float16 half8  __attribute__((ext_vector_type(8)));
typedef _Float16 half16 __attribute__((ext_vector_type(16)));
typedef float    float8 __attribute__((ext_vector_type(8)));

// ---------------- weight f32 -> f16 cast ----------------
__global__ void k_cast_h(const float* __restrict__ s, _Float16* __restrict__ d, int n) {
    int i = blockIdx.x * blockDim.x + threadIdx.x;
    if (i < n) d[i] = (_Float16)s[i];
}

// ---------------- BN scale/shift precompute ----------------
__global__ void k_bn_prep(const float* __restrict__ g, const float* __restrict__ b,
                          const float* __restrict__ m, const float* __restrict__ v,
                          float* __restrict__ sc, float* __restrict__ sh) {
    int o = threadIdx.x;
    float s = g[o] * rsqrtf(v[o] + EPSF);
    sc[o] = s;
    sh[o] = b[o] - m[o] * s;
}

// ---------------- LayerNorm over C=256 + chunk split ----------------
// x:(B,C,H,W) -> xs:(16, L, 64) row = k*4+b ; writes f32 (skip path) + f16 (GEMM A)
__global__ void k_ln_split(const float* __restrict__ x, const float* __restrict__ g,
                           const float* __restrict__ beta, float* __restrict__ xs32,
                           _Float16* __restrict__ xs16) {
    int t = blockIdx.x * blockDim.x + threadIdx.x;   // token over B*L
    if (t >= Bb * Lt) return;
    int b = t / Lt, l = t - b * Lt;
    const float* px = x + (size_t)b * Cc * Lt + l;   // channel stride = L
    float s = 0.f, ss = 0.f;
    for (int c = 0; c < Cc; ++c) { float v = px[(size_t)c * Lt]; s += v; ss += v * v; }
    float mean = s * (1.f / Cc);
    float var  = ss * (1.f / Cc) - mean * mean;
    float inv  = rsqrtf(var + EPSF);
    for (int c = 0; c < Cc; ++c) {
        float v = (px[(size_t)c * Lt] - mean) * inv * g[c] + beta[c];
        int k = c >> 6, d = c & 63;
        size_t idx = ((size_t)(k * 4 + b) * Lt + l) * DMc + d;
        xs32[idx] = v;
        xs16[idx] = (_Float16)v;
    }
}

// ---------------- generic fused WMMA GEMM, 2x4 register-blocked ----------------
// Wave computes a 32x64 output block (2 M-tiles x 4 N-tiles): A frags reused 4x,
// B frags reused 2x.  A: (M,K) f16 row-major.  W: (N,K) f16 row-major.
// act: 0 none, 1 exact GELU, 2 SiLU.  out_bcl: store as [row/L][col][row%L].
__global__ void k_gemm16(const _Float16* __restrict__ A, const _Float16* __restrict__ W,
                         const float* __restrict__ bias,
                         const float* __restrict__ scale, const float* __restrict__ shift,
                         const float* __restrict__ addmat, const float* __restrict__ addscale,
                         float* __restrict__ outf, _Float16* __restrict__ outh,
                         int M, int N, int K, int act, int out_bcl) {
    int wid  = (blockIdx.x * blockDim.x + threadIdx.x) >> 5;
    int lane = threadIdx.x & 31;
    int tn = N >> 6;                       // n-blocks of 64
    if (wid >= (M >> 5) * tn) return;      // m-blocks of 32
    int tmb = wid / tn, tnb = wid - tmb * tn;
    int m0 = tmb << 5, n0 = tnb << 6;
    int r  = lane & 15;      // A row / W row (=out col) within tile
    int hg = lane >> 4;      // K-half selector: k = 16*hg + s for A and W (contiguous order)
    const _Float16* arow0 = A + (size_t)(m0 + r) * K + hg * 16;
    const _Float16* arow1 = arow0 + (size_t)16 * K;
    const _Float16* wrow  = W + (size_t)(n0 + r) * K + hg * 16;
    size_t wstep = (size_t)16 * K;         // next n-tile's rows
    float8 c[2][4] = {};
    for (int k0 = 0; k0 < K; k0 += 32) {
        half8 a0lo = *(const half8*)(arow0 + k0);
        half8 a0hi = *(const half8*)(arow0 + k0 + 8);
        half8 a1lo = *(const half8*)(arow1 + k0);
        half8 a1hi = *(const half8*)(arow1 + k0 + 8);
        half16 af0 = __builtin_shufflevector(a0lo, a0hi, 0,1,2,3,4,5,6,7,8,9,10,11,12,13,14,15);
        half16 af1 = __builtin_shufflevector(a1lo, a1hi, 0,1,2,3,4,5,6,7,8,9,10,11,12,13,14,15);
#pragma unroll
        for (int j = 0; j < 4; ++j) {
            half16 bf = *(const half16*)(wrow + (size_t)j * wstep + k0);
            c[0][j] = __builtin_amdgcn_wmma_f32_16x16x32_f16(false, af0, false, bf,
                                                             (short)0, c[0][j], false, false);
            c[1][j] = __builtin_amdgcn_wmma_f32_16x16x32_f16(false, af1, false, bf,
                                                             (short)0, c[1][j], false, false);
        }
    }
    float asc = addmat ? addscale[0] : 0.f;
#pragma unroll
    for (int j = 0; j < 4; ++j) {
        int col = n0 + (j << 4) + r;
        float bs = bias  ? bias[col]  : 0.f;
        float sc = scale ? scale[col] : 1.f;
        float sh = shift ? shift[col] : 0.f;
#pragma unroll
        for (int im = 0; im < 2; ++im) {
#pragma unroll
            for (int v = 0; v < 8; ++v) {
                int row = m0 + (im << 4) + v + hg * 8;
                float y = c[im][j][v] + bs;
                if (scale)  y = y * sc + sh;
                if (addmat) y += asc * addmat[(size_t)row * N + col];
                if (act == 1)      y = 0.5f * y * (1.f + erff(y * 0.70710678f));
                else if (act == 2) y = y / (1.f + __expf(-y));
                size_t idx = out_bcl ? ((size_t)(row >> 12) * N + col) * Lt + (row & (Lt - 1))
                                     : (size_t)row * N + col;
                if (outf) outf[idx] = y;
                if (outh) outh[idx] = (_Float16)y;
            }
        }
    }
}

// ---------------- depthwise causal conv (K=4) + bias + SiLU ----------------
__global__ void k_conv_silu(const float* __restrict__ xz, const float* __restrict__ wc,
                            const float* __restrict__ bc, float* __restrict__ xca) {
    size_t i = (size_t)blockIdx.x * blockDim.x + threadIdx.x;
    if (i >= (size_t)NTt * DIc) return;
    int d = (int)(i & (DIc - 1));
    size_t t = i >> 7;                 // nb*L + l
    int l = (int)(t & (Lt - 1));
    size_t base = (t - l) * Cc;        // (nb*L)*256
    float acc = bc[d];
#pragma unroll
    for (int j = 0; j < KCc; ++j) {
        int ll = l - (KCc - 1) + j;
        if (ll >= 0) acc += wc[d * KCc + j] * xz[base + (size_t)ll * Cc + d];
    }
    acc = acc / (1.f + __expf(-acc));  // SiLU
    xca[i] = acc;
}

// ---------------- x_dbl = xc_act @ W_xproj^T (36 outs, K=128) ----------------
__global__ void k_xdbl(const float* __restrict__ xca, const float* __restrict__ Wx,
                       float* __restrict__ xd) {
    size_t i = (size_t)blockIdx.x * blockDim.x + threadIdx.x;
    if (i >= (size_t)NTt * 36) return;
    int j = (int)(i % 36);
    size_t t = i / 36;
    const float* a = xca + t * DIc;
    const float* w = Wx + (size_t)j * DIc;
    float acc = 0.f;
    for (int k = 0; k < DIc; ++k) acc += a[k] * w[k];
    xd[i] = acc;
}

// ---------------- selective scan: h[16] in regs, token row via LDS broadcast ----------------
// block = one sequence (nb in 0..15), 128 threads = channels. Fuses dt/softplus,
// state update, C-contraction, +D*xc, SiLU(z) gate; writes f16 GEMM input.
// Prefetch (global_prefetch_b8) a few steps ahead to hide latency on the serial path.
__global__ void __launch_bounds__(128) k_scan(
        const float* __restrict__ xd, const float* __restrict__ xca,
        const float* __restrict__ xz, const float* __restrict__ Wdt,
        const float* __restrict__ bdt, const float* __restrict__ Alog,
        const float* __restrict__ Dp, _Float16* __restrict__ yg) {
    __shared__ float row[64];
    int nb = blockIdx.x, d = threadIdx.x;
    float An[NSc];
#pragma unroll
    for (int n = 0; n < NSc; ++n) An[n] = -__expf(Alog[d * NSc + n]);
    float h[NSc];
#pragma unroll
    for (int n = 0; n < NSc; ++n) h[n] = 0.f;
    float w0 = Wdt[d*4+0], w1 = Wdt[d*4+1], w2 = Wdt[d*4+2], w3 = Wdt[d*4+3];
    float bd = bdt[d], Dd = Dp[d];
    size_t t0 = (size_t)nb * Lt;
    for (int l = 0; l < Lt; ++l) {
        size_t t = t0 + l;
        if (l + 8 < Lt) {
            __builtin_prefetch(xd + (t + 8) * 36, 0, 1);
            __builtin_prefetch(xca + (t + 8) * DIc + d, 0, 1);
            __builtin_prefetch(xz + (t + 8) * Cc + DIc + d, 0, 1);
        }
        if (d < 36) row[d] = xd[t * 36 + d];
        __syncthreads();
        float dtr = bd + row[0]*w0 + row[1]*w1 + row[2]*w2 + row[3]*w3;
        float dt  = (dtr > 20.f) ? dtr : log1pf(__expf(dtr));   // softplus
        float xc  = xca[t * DIc + d];
        float acc = 0.f;
#pragma unroll
        for (int n = 0; n < NSc; ++n) {
            float a = __expf(dt * An[n]);
            h[n] = a * h[n] + dt * row[4 + n] * xc;   // B_n
            acc += h[n] * row[20 + n];                // C_n
        }
        float z = xz[t * Cc + DIc + d];
        float y = (acc + Dd * xc) * (z / (1.f + __expf(-z)));
        yg[t * DIc + d] = (_Float16)y;
        __syncthreads();
    }
}

// ---------------- LayerNorm over DM=64, writes f16 for fc1 ----------------
__global__ void k_ln64(const float* __restrict__ ym, const float* __restrict__ g,
                       const float* __restrict__ b, _Float16* __restrict__ yn) {
    int t = blockIdx.x * blockDim.x + threadIdx.x;
    if (t >= NTt) return;
    const float* p = ym + (size_t)t * DMc;
    float s = 0.f, ss = 0.f;
    for (int c = 0; c < DMc; ++c) { float v = p[c]; s += v; ss += v * v; }
    float mean = s * (1.f / DMc);
    float var  = ss * (1.f / DMc) - mean * mean;
    float inv  = rsqrtf(var + EPSF);
    for (int c = 0; c < DMc; ++c)
        yn[(size_t)t * DMc + c] = (_Float16)((p[c] - mean) * inv * g[c] + b[c]);
}

// ---------------- re-interleave: ym2(16,L,64) -> xi(B*L, 256), c = d*4 + k ----------------
__global__ void k_repack(const float* __restrict__ ym2, _Float16* __restrict__ xi) {
    size_t i = (size_t)blockIdx.x * blockDim.x + threadIdx.x;
    if (i >= (size_t)Bb * Lt * Cc) return;
    int c = (int)(i & 255);
    int l = (int)((i >> 8) & (Lt - 1));
    int b = (int)(i >> 20);
    int dd = c >> 2, k = c & 3;                               // transpose(1,3,0,2): c = d*4 + k
    size_t src = ((size_t)(k * 4 + b) * Lt + l) * DMc + dd;   // mamba row = k*4 + b
    xi[i] = (_Float16)ym2[src];
}

extern "C" void kernel_launch(void* const* d_in, const int* in_sizes, int n_in,
                              void* d_out, int out_size, void* d_ws, size_t ws_size,
                              hipStream_t stream) {
    (void)in_sizes; (void)n_in; (void)out_size; (void)ws_size;
    const float* x      = (const float*)d_in[0];
    const float* g_norm = (const float*)d_in[1];
    const float* b_norm = (const float*)d_in[2];
    const float* g_n1   = (const float*)d_in[3];
    const float* b_n1   = (const float*)d_in[4];
    const float* W_in   = (const float*)d_in[5];
    const float* W_conv = (const float*)d_in[6];
    const float* b_conv = (const float*)d_in[7];
    const float* W_xp   = (const float*)d_in[8];
    const float* W_dt   = (const float*)d_in[9];
    const float* b_dt   = (const float*)d_in[10];
    const float* A_log  = (const float*)d_in[11];
    const float* D_par  = (const float*)d_in[12];
    const float* W_outp = (const float*)d_in[13];
    const float* skip_s = (const float*)d_in[14];
    const float* W_fc1  = (const float*)d_in[15];
    const float* b_fc1  = (const float*)d_in[16];
    const float* W_fc2  = (const float*)d_in[17];
    const float* b_fc2  = (const float*)d_in[18];
    const float* W_out  = (const float*)d_in[19];
    const float* bn_g   = (const float*)d_in[20];
    const float* bn_b   = (const float*)d_in[21];
    const float* bn_m   = (const float*)d_in[22];
    const float* bn_v   = (const float*)d_in[23];

    char* ws = (char*)d_ws;
    size_t off = 0;
    auto alloc = [&](size_t bytes) { size_t r = off; off += (bytes + 255) & ~(size_t)255; return r; };
    float*    xs32 = (float*)   (ws + alloc((size_t)NTt * DMc * 4));
    _Float16* xs16 = (_Float16*)(ws + alloc((size_t)NTt * DMc * 2));
    float*    xz   = (float*)   (ws + alloc((size_t)NTt * Cc  * 4));
    float*    xca  = (float*)   (ws + alloc((size_t)NTt * DIc * 4));
    float*    xd   = (float*)   (ws + alloc((size_t)NTt * 36  * 4));
    _Float16* yg   = (_Float16*)(ws + alloc((size_t)NTt * DIc * 2));
    float*    ym   = (float*)   (ws + alloc((size_t)NTt * DMc * 4));
    _Float16* yn   = (_Float16*)(ws + alloc((size_t)NTt * DMc * 2));
    _Float16* f1   = (_Float16*)(ws + alloc((size_t)NTt * Cc  * 2));
    float*    ym2  = (float*)   (ws + alloc((size_t)NTt * DMc * 4));
    _Float16* xi   = (_Float16*)(ws + alloc((size_t)Bb * Lt * Cc * 2));
    _Float16* Win16  = (_Float16*)(ws + alloc(256 * 64  * 2));
    _Float16* Wop16  = (_Float16*)(ws + alloc(64  * 128 * 2));
    _Float16* Wf116  = (_Float16*)(ws + alloc(256 * 64  * 2));
    _Float16* Wf216  = (_Float16*)(ws + alloc(64  * 256 * 2));
    _Float16* Wout16 = (_Float16*)(ws + alloc(256 * 256 * 2));
    float* bnsc = (float*)(ws + alloc(256 * 4));
    float* bnsh = (float*)(ws + alloc(256 * 4));

    auto cast = [&](const float* s, _Float16* d, int n) {
        k_cast_h<<<(n + 255) / 256, 256, 0, stream>>>(s, d, n);
    };
    auto gemm = [&](const _Float16* A, const _Float16* W, const float* bias,
                    const float* sc, const float* sh, const float* add, const float* ascl,
                    float* of, _Float16* oh, int M, int N, int K, int act, int bcl) {
        int waves  = (M / 32) * (N / 64);        // 2x4 blocked tiles per wave
        int blocks = (waves * 32 + 255) / 256;
        k_gemm16<<<blocks, 256, 0, stream>>>(A, W, bias, sc, sh, add, ascl,
                                             of, oh, M, N, K, act, bcl);
    };

    cast(W_in,   Win16,  256 * 64);
    cast(W_outp, Wop16,  64 * 128);
    cast(W_fc1,  Wf116,  256 * 64);
    cast(W_fc2,  Wf216,  64 * 256);
    cast(W_out,  Wout16, 256 * 256);
    k_bn_prep<<<1, 256, 0, stream>>>(bn_g, bn_b, bn_m, bn_v, bnsc, bnsh);

    // LN(C) + split
    k_ln_split<<<(Bb * Lt + 255) / 256, 256, 0, stream>>>(x, g_norm, b_norm, xs32, xs16);
    // in-proj: (65536,64) x (64->256)
    gemm(xs16, Win16, nullptr, nullptr, nullptr, nullptr, nullptr,
         xz, nullptr, NTt, 256, 64, 0, 0);
    // depthwise causal conv + SiLU on xc half
    k_conv_silu<<<(int)(((size_t)NTt * DIc + 255) / 256), 256, 0, stream>>>(xz, W_conv, b_conv, xca);
    // x_dbl (dt-rank + B + C projections)
    k_xdbl<<<(int)(((size_t)NTt * 36 + 255) / 256), 256, 0, stream>>>(xca, W_xp, xd);
    // selective scan + gate
    k_scan<<<NBc, 128, 0, stream>>>(xd, xca, xz, W_dt, b_dt, A_log, D_par, yg);
    // out-proj: (65536,128) x (128->64)
    gemm(yg, Wop16, nullptr, nullptr, nullptr, nullptr, nullptr,
         ym, nullptr, NTt, 64, 128, 0, 0);
    // LN(64)
    k_ln64<<<(NTt + 255) / 256, 256, 0, stream>>>(ym, g_n1, b_n1, yn);
    // fc1 + GELU (f16 out for fc2)
    gemm(yn, Wf116, b_fc1, nullptr, nullptr, nullptr, nullptr,
         nullptr, f1, NTt, 256, 64, 1, 0);
    // fc2 + bias + skip_scale * xs
    gemm(f1, Wf216, b_fc2, nullptr, nullptr, xs32, skip_s,
         ym2, nullptr, NTt, 64, 256, 0, 0);
    // channel re-interleave for 1x1 conv
    k_repack<<<(int)(((size_t)Bb * Lt * Cc + 255) / 256), 256, 0, stream>>>(ym2, xi);
    // W_out GEMM + BN affine + SiLU, stored as (B,C,H,W)
    gemm(xi, Wout16, nullptr, bnsc, bnsh, nullptr, nullptr,
         (float*)d_out, nullptr, Bb * Lt, 256, 256, 2, 1);
}
